// ColumnParallelLinearWithPagedLoRA_28097676051188
// MI455X (gfx1250) — matrix operand order repaired and checked by reading, not compiled, not run on hardware
//
#include <hip/hip_runtime.h>

// ---- problem sizes (fixed by setup_inputs) ----
constexpr int Tdim = 4096;   // tokens
constexpr int Hdim = 4096;   // hidden
constexpr int Odim = 4096;   // out features
constexpr int Rr   = 16;     // lora rank
constexpr int Sseg = 8;      // segments

// ---- tiling ----
constexpr int BM = 128;
constexpr int BN = 128;
constexpr int KT = 32;                 // K per WMMA step (bf16)
constexpr int KTILES = Hdim / KT;      // 128

typedef __attribute__((ext_vector_type(16))) __bf16 v16bf;
typedef __attribute__((ext_vector_type(8)))  float  v8f;
typedef int v4i __attribute__((vector_size(16)));    // matches builtin's V4i

#if __has_builtin(__builtin_amdgcn_global_load_async_to_lds_b128) && \
    __has_builtin(__builtin_amdgcn_s_wait_asynccnt)
#define USE_ASYNC 1
#else
#define USE_ASYNC 0
#endif

#if USE_ASYNC
#define WAIT_ASYNC(n) __builtin_amdgcn_s_wait_asynccnt(n)
__device__ __forceinline__ void async_cp16(const float* g, float* l) {
  __builtin_amdgcn_global_load_async_to_lds_b128(
      (__attribute__((address_space(1))) v4i*)(g),
      (__attribute__((address_space(3))) v4i*)(l),
      0, 0);
}
#else
#define WAIT_ASYNC(n) do {} while (0)
__device__ __forceinline__ void async_cp16(const float* g, float* l) {
  *(float4*)l = *(const float4*)g;
}
#endif

__device__ __forceinline__ void put8(v16bf& d, int base, float4 a, float4 b) {
  d[base + 0] = (__bf16)a.x; d[base + 1] = (__bf16)a.y;
  d[base + 2] = (__bf16)a.z; d[base + 3] = (__bf16)a.w;
  d[base + 4] = (__bf16)b.x; d[base + 5] = (__bf16)b.y;
  d[base + 6] = (__bf16)b.z; d[base + 7] = (__bf16)b.w;
}

__global__ __launch_bounds__(256)
void cpl_paged_lora_gemm(const float* __restrict__ x,
                         const float* __restrict__ wgt,
                         const float* __restrict__ bias,
                         const float* __restrict__ la,
                         const float* __restrict__ lb,
                         const long long* __restrict__ idx,
                         float* __restrict__ out)
{
  __shared__ __align__(16) float xs [2][BM][KT + 4];   // x tile (f32)
  __shared__ __align__(16) float wsb[2][BN][KT + 4];   // W tile (f32)
  __shared__ __align__(16) float as_[2][KT][Rr + 4];   // lora_a tile (f32)
  __shared__ __align__(16) float us [BM][Rr + 1];      // u = x@A[l] for this block

  const int tid    = threadIdx.x;
  const int lane   = tid & 31;
  const int wv     = tid >> 5;
  const int wave_m = wv >> 2;               // 0..1  -> 64 rows each
  const int wave_n = wv & 3;                // 0..3  -> 32 cols each
  const int m0 = blockIdx.y * BM;
  const int n0 = blockIdx.x * BN;

  // Segment LoRA id: segments are (T/S)=512-aligned, block of 128 rows is
  // fully inside one segment -> single id per block (searchsorted semantics).
  int lsel = 0;
  for (int i = 0; i < Sseg; ++i) {
    if (idx[2 * i] <= (long long)m0) lsel = (int)idx[2 * i + 1];
  }
  const float* __restrict__ laL = la + (size_t)lsel * Hdim * Rr;
  const float* __restrict__ lbL = lb + (size_t)lsel * Rr * Odim;

  v8f acc[4][2] = {};
  v8f uacc[4]   = {};

  auto issue = [&](int kt, int buf) {
    const int k0 = kt * KT;
#pragma unroll
    for (int i = 0; i < 4; ++i) {            // x: 128x32 f32 = 1024 chunks/16B
      int c = tid + 256 * i;
      int row = c >> 3, ch = c & 7;
      async_cp16(x + (size_t)(m0 + row) * Hdim + k0 + ch * 4,
                 &xs[buf][row][ch * 4]);
    }
#pragma unroll
    for (int i = 0; i < 4; ++i) {            // W: 128x32 f32
      int c = tid + 256 * i;
      int row = c >> 3, ch = c & 7;
      async_cp16(wgt + (size_t)(n0 + row) * Hdim + k0 + ch * 4,
                 &wsb[buf][row][ch * 4]);
    }
    if (tid < 128) {                         // A[l]: 32x16 f32 = 128 chunks
      int k = tid >> 2, ch = tid & 3;
      async_cp16(laL + (size_t)(k0 + k) * Rr + ch * 4,
                 &as_[buf][k][ch * 4]);
    }
  };

  const int lh   = lane >> 4;      // lane half
  const int ll   = lane & 15;
  const int off1 = lh * 8;         // A-frag first K run (ISA 7.12.2)
  const int kb2  = lh * 16;        // B-frag K base

  auto compute = [&](int buf) {
    v16bf afr[4];
#pragma unroll
    for (int mt = 0; mt < 4; ++mt) {         // A: row ll, 16 contiguous K @kb2
      const float* rp = &xs[buf][wave_m * 64 + mt * 16 + ll][0];
      float4 a0 = *(const float4*)(rp + kb2);
      float4 a1 = *(const float4*)(rp + kb2 + 4);
      float4 a2 = *(const float4*)(rp + kb2 + 8);
      float4 a3 = *(const float4*)(rp + kb2 + 12);
      put8(afr[mt], 0, a0, a1);
      put8(afr[mt], 8, a2, a3);
    }
    v16bf bfr[2];
#pragma unroll
    for (int nt = 0; nt < 2; ++nt) {         // B: col ll, 16 contiguous K @kb2
      const float* rp = &wsb[buf][wave_n * 32 + nt * 16 + ll][kb2];
      float4 b0 = *(const float4*)(rp + 0);
      float4 b1 = *(const float4*)(rp + 4);
      float4 b2 = *(const float4*)(rp + 8);
      float4 b3 = *(const float4*)(rp + 12);
      put8(bfr[nt], 0, b0, b1);
      put8(bfr[nt], 8, b2, b3);
    }
#pragma unroll
    for (int mt = 0; mt < 4; ++mt)
#pragma unroll
      for (int nt = 0; nt < 2; ++nt)
        acc[mt][nt] = __builtin_amdgcn_wmma_f32_16x16x32_bf16(
            false, afr[mt], false, bfr[nt], (short)0, acc[mt][nt], false, false);

    if (wave_n == 0) {                       // u += x_tile @ A[l]_tile (N=R=16)
      v16bf ub;
#pragma unroll
      for (int i = 0; i < 16; ++i) ub[i] = (__bf16)as_[buf][kb2 + i][ll];
#pragma unroll
      for (int mt = 0; mt < 4; ++mt)
        uacc[mt] = __builtin_amdgcn_wmma_f32_16x16x32_bf16(
            false, afr[mt], false, ub, (short)0, uacc[mt], false, false);
    }
  };

  issue(0, 0);
  for (int kt = 0; kt < KTILES; ++kt) {
    if (kt + 1 < KTILES) {
      issue(kt + 1, (kt + 1) & 1);
      if (wv < 4) { WAIT_ASYNC(9); } else { WAIT_ASYNC(8); }  // keep next stage in flight
    } else {
      WAIT_ASYNC(0);
    }
    __syncthreads();
    compute(kt & 1);
    __syncthreads();
  }

  // ---- share u across the block ----
  if (wave_n == 0) {
#pragma unroll
    for (int mt = 0; mt < 4; ++mt)
#pragma unroll
      for (int i = 0; i < 8; ++i)
        us[wave_m * 64 + mt * 16 + lh * 8 + i][ll] = uacc[mt][i];
  }
  __syncthreads();

  // ---- acc += u @ B[l]  (K=16 zero-padded to 32, one WMMA per tile) ----
  v16bf bcf[2];
#pragma unroll
  for (int nt = 0; nt < 2; ++nt) {
    v16bf b;
#pragma unroll
    for (int i = 0; i < 16; ++i) b[i] = (__bf16)0.0f;
    if (lh == 0) {                           // K 0..15 real; upper half zero
      const float* bp = lbL + (size_t)(n0 + wave_n * 32 + nt * 16 + ll);
#pragma unroll
      for (int i = 0; i < 16; ++i) b[i] = (__bf16)bp[(size_t)i * Odim];
    }
    bcf[nt] = b;
  }
#pragma unroll
  for (int mt = 0; mt < 4; ++mt) {
    v16bf ua;
#pragma unroll
    for (int i = 0; i < 16; ++i) ua[i] = (__bf16)0.0f;
    const float* up = &us[wave_m * 64 + mt * 16 + ll][0];
#pragma unroll
    for (int j = 0; j < 16; ++j) ua[j] = lh ? (__bf16)0.0f : (__bf16)up[j];  // K=0..15 on lh 0
#pragma unroll
    for (int nt = 0; nt < 2; ++nt)
      acc[mt][nt] = __builtin_amdgcn_wmma_f32_16x16x32_bf16(
          false, ua, false, bcf[nt], (short)0, acc[mt][nt], false, false);
  }

  // ---- bias + store (C layout: elem i -> M = i + 8*lh, N = ll) ----
#pragma unroll
  for (int nt = 0; nt < 2; ++nt) {
    const int col = n0 + wave_n * 32 + nt * 16 + ll;
    const float bv = bias[col];
#pragma unroll
    for (int mt = 0; mt < 4; ++mt) {
      const int rowb = m0 + wave_m * 64 + mt * 16 + lh * 8;
#pragma unroll
      for (int i = 0; i < 8; ++i)
        out[(size_t)(rowb + i) * Odim + col] = acc[mt][nt][i] + bv;
    }
  }
}

extern "C" void kernel_launch(void* const* d_in, const int* in_sizes, int n_in,
                              void* d_out, int out_size, void* d_ws, size_t ws_size,
                              hipStream_t stream) {
  (void)in_sizes; (void)n_in; (void)out_size; (void)d_ws; (void)ws_size;
  const float*     x   = (const float*)d_in[0];
  const float*     w   = (const float*)d_in[1];
  const float*     b   = (const float*)d_in[2];
  const float*     la  = (const float*)d_in[3];
  const float*     lb  = (const float*)d_in[4];
  const long long* idx = (const long long*)d_in[5];
  float*           out = (float*)d_out;

  dim3 grid(Odim / BN, Tdim / BM);   // 32 x 32 = 1024 blocks
  cpl_paged_lora_gemm<<<grid, dim3(256), 0, stream>>>(x, w, b, la, lb, idx, out);
}
